// VanillaMOEDecoder_3453153706570
// MI455X (gfx1250) — matrix operand order repaired and checked by reading, not compiled, not run on hardware
//
#include <hip/hip_runtime.h>
#include <hip/hip_bf16.h>

// ---------------------------------------------------------------------------
// VanillaMOEDecoder for MI455X (gfx1250, wave32):
//   WMMA bf16 (f32 accum) GEMMs + flash attention + top-1 MoE routing.
//   Async global->LDS (ASYNCcnt) staging for bf16 tiles.
// ---------------------------------------------------------------------------

typedef __attribute__((ext_vector_type(16))) __bf16 v16bf;
typedef __attribute__((ext_vector_type(8)))  float  v8f;

#define NB  8192   // B*S tokens
#define DD  512
#define FFD 2048
#define NE  8
#define SS  2048
#define NH  8
#define HDD 64
#define MAXTILES 144

__device__ __forceinline__ v8f wmma_bf16(v16bf a, v16bf b, v8f c) {
  return __builtin_amdgcn_wmma_f32_16x16x32_bf16(
      /*neg_a=*/false, a, /*neg_b=*/false, b,
      /*c_mod=*/(short)0, c, /*reuse_a=*/false, /*reuse_b=*/false);
}

// Async 16-byte global -> LDS copy (tracked by ASYNCcnt).
__device__ __forceinline__ void async_ld_b128(const void* lds_ptr, const void* gptr) {
  uint32_t lds_off = (uint32_t)(uintptr_t)lds_ptr;   // low 32 bits = LDS byte addr
  asm volatile("global_load_async_to_lds_b128 %0, %1, off"
               :: "v"(lds_off), "v"(gptr) : "memory");
}
__device__ __forceinline__ void wait_async0() {
  asm volatile("s_wait_asynccnt 0x0" ::: "memory");
}

// Pack two fp32 into one dword of bf16 (lo->bits[15:0], hi->bits[31:16]).
__device__ __forceinline__ uint32_t pack_bf16(float lo, float hi) {
  __bf16 l = (__bf16)lo, h = (__bf16)hi;
  unsigned short ls, hs;
  __builtin_memcpy(&ls, &l, 2);
  __builtin_memcpy(&hs, &h, 2);
  return ((uint32_t)hs << 16) | ls;
}

// A fragment (16x32 bf16), base at (row0,k0) of a row-major LDS tile.
__device__ __forceinline__ v16bf frag_a(const __bf16* base, int lda, int lane) {
  const __bf16* p = base + (lane & 15) * lda;
  const int h = (lane >> 4) * 8;
  v16bf a;
#pragma unroll
  for (int j = 0; j < 8; ++j) a[j] = p[h + j];
#pragma unroll
  for (int j = 0; j < 8; ++j) a[8 + j] = p[16 + h + j];
  return a;
}

// B fragment (32x16 bf16) from LDS tile stored transposed: Bs[n][k].
__device__ __forceinline__ v16bf frag_b(const __bf16* base, int ldb, int lane) {
  const __bf16* p = base + (lane & 15) * ldb + (lane >> 4) * 8;
  v16bf b;
#pragma unroll
  for (int j = 0; j < 8; ++j) b[j] = p[j];
#pragma unroll
  for (int j = 0; j < 8; ++j) b[8 + j] = p[16 + j];
  return b;
}

__device__ __forceinline__ float gelu_tanh(float x) {
  float x3 = x * x * x;
  float u = 0.7978845608028654f * (x + 0.044715f * x3);
  return 0.5f * x * (1.f + tanhf(u));
}

// ---------------------------------------------------------------------------
// GEMM: C[M,N] = A[M,K] @ W[K,N] + bias[N]; A,W fp32; OutT = float or __bf16.
// 64x64 tile, BK=32, 256 threads = 8 waves (4x2), each wave 16x32.
// ---------------------------------------------------------------------------
template <typename OutT>
__global__ __launch_bounds__(256) void gemm_kernel(
    const float* __restrict__ A, const float* __restrict__ W,
    const float* __restrict__ bias, OutT* __restrict__ C,
    int M, int N, int K) {
  __shared__ __align__(16) __bf16 As[64][32];
  __shared__ __align__(16) __bf16 Bs[64][32];
  const int t = threadIdx.x;
  const int lane = t & 31;
  const int w = t >> 5;
  const int wr = w >> 1, wc = w & 1;
  const int m0 = blockIdx.x * 64;
  const int n0 = blockIdx.y * 64;
  v8f c0 = {}; v8f c1 = {};
  for (int k0 = 0; k0 < K; k0 += 32) {
    __syncthreads();
    if (k0 + 32 < K)   // prefetch next weight tile (global_prefetch_b8)
      __builtin_prefetch(&W[(size_t)(k0 + 32 + (t >> 6)) * N + n0 + (t & 63)], 0, 3);
#pragma unroll
    for (int r = 0; r < 4; ++r) {            // A tile 64x32, packed cvt stores
      int i = t + r * 256;                    // 0..1023 pairs
      int row = i >> 4, cp = i & 15;
      const float* src = &A[(size_t)(m0 + row) * K + k0 + cp * 2];
      ((uint32_t*)As)[row * 16 + cp] = pack_bf16(src[0], src[1]);
    }
#pragma unroll
    for (int r = 0; r < 4; ++r) {            // W tile 32x64 -> Bs[n][k] packed along k
      int i = t + r * 256;
      int n = i & 63, kp = i >> 6;            // kp = 0..15
      size_t gi = (size_t)(k0 + kp * 2) * N + n0 + n;
      ((uint32_t*)Bs)[n * 16 + kp] = pack_bf16(W[gi], W[gi + N]);
    }
    __syncthreads();
    v16bf a  = frag_a(&As[wr * 16][0], 32, lane);
    v16bf b0 = frag_b(&Bs[wc * 32][0], 32, lane);
    v16bf b1 = frag_b(&Bs[wc * 32 + 16][0], 32, lane);
    c0 = wmma_bf16(a, b0, c0);
    c1 = wmma_bf16(a, b1, c1);
  }
  const int nlo = n0 + wc * 32 + (lane & 15);
  const int mrow = m0 + wr * 16 + ((lane >> 4) << 3);
#pragma unroll
  for (int j = 0; j < 8; ++j) {
    C[(size_t)(mrow + j) * N + nlo]      = (OutT)(c0[j] + bias[nlo]);
    C[(size_t)(mrow + j) * N + nlo + 16] = (OutT)(c1[j] + bias[nlo + 16]);
  }
}

// ---------------------------------------------------------------------------
// Flash attention: grid (S/64, B*H). Q/K/V are bf16 [B,S,D]; O is fp32.
// Q tile + K chunks staged via async global->LDS; V transposed manually.
// ---------------------------------------------------------------------------
__global__ __launch_bounds__(256) void flash_kernel(
    const __bf16* __restrict__ Q, const __bf16* __restrict__ K,
    const __bf16* __restrict__ V, float* __restrict__ O) {
  __shared__ __align__(16) __bf16 Qs[64][64];
  __shared__ __align__(16) __bf16 Ks[64][64];   // Ks[key][hd] == B tile (n=key,k=hd)
  __shared__ __align__(16) __bf16 Vs[64][64];   // Vs[hd][key] == B tile (n=hd,k=key)
  __shared__ float  Sc[64][64];
  __shared__ __bf16 Ps[64][64];
  __shared__ float  m_s[64], l_s[64], cor_s[64];

  const int t = threadIdx.x, lane = t & 31, w = t >> 5;
  const int wr = w >> 1, wc = w & 1;
  const int q0 = blockIdx.x * 64;
  const int bh = blockIdx.y;
  const int bb = bh >> 3, hh = bh & 7;
  const size_t base = (size_t)bb * SS * DD + (size_t)hh * HDD;

#pragma unroll
  for (int r = 0; r < 2; ++r) {            // Q tile: 64 rows x 128B = 512 x 16B
    int i = t + r * 256;
    int row = i >> 3, seg = i & 7;
    async_ld_b128(&Qs[row][seg * 8], Q + base + (size_t)(q0 + row) * DD + seg * 8);
  }
  if (t < 64) { m_s[t] = -1e30f; l_s[t] = 0.f; }
  v8f o0 = {}; v8f o1 = {};

  const int sn = wc * 32 + (lane & 15);
  const int sm = wr * 16 + ((lane >> 4) << 3);

  for (int kc = 0; kc < SS; kc += 64) {
    __syncthreads();
#pragma unroll
    for (int r = 0; r < 2; ++r) {          // K chunk: async copy, layout matches Ks
      int i = t + r * 256;
      int row = i >> 3, seg = i & 7;
      async_ld_b128(&Ks[row][seg * 8], K + base + (size_t)(kc + row) * DD + seg * 8);
    }
#pragma unroll
    for (int r = 0; r < 16; ++r) {         // V chunk transposed (bf16 copy)
      int i = t + r * 256;
      int row = i >> 6, col = i & 63;
      Vs[col][row] = V[base + (size_t)(kc + row) * DD + col];
    }
    wait_async0();
    __syncthreads();

    v8f s0 = {}; v8f s1 = {};
#pragma unroll
    for (int kk = 0; kk < 64; kk += 32) {  // S = Q @ K^T
      v16bf a  = frag_a(&Qs[wr * 16][kk], 64, lane);
      v16bf b0 = frag_b(&Ks[wc * 32][kk], 64, lane);
      v16bf b1 = frag_b(&Ks[wc * 32 + 16][kk], 64, lane);
      s0 = wmma_bf16(a, b0, s0);
      s1 = wmma_bf16(a, b1, s1);
    }
#pragma unroll
    for (int j = 0; j < 8; ++j) {
      Sc[sm + j][sn]      = s0[j] * 0.125f;   // 1/sqrt(64)
      Sc[sm + j][sn + 16] = s1[j] * 0.125f;
    }
    __syncthreads();

    if (t < 64) {                           // online softmax, one row per thread
      float mold = m_s[t];
      float mc = mold;
      for (int k2 = 0; k2 < 64; ++k2) mc = fmaxf(mc, Sc[t][k2]);
      float cor = __expf(mold - mc);
      float sum = 0.f;
      for (int k2 = 0; k2 < 64; ++k2) {
        float p = __expf(Sc[t][k2] - mc);
        Ps[t][k2] = (__bf16)p;
        sum += p;
      }
      l_s[t] = l_s[t] * cor + sum;
      m_s[t] = mc;
      cor_s[t] = cor;
    }
    __syncthreads();

#pragma unroll
    for (int j = 0; j < 8; ++j) {           // rescale running O
      float cr = cor_s[sm + j];
      o0[j] *= cr; o1[j] *= cr;
    }
#pragma unroll
    for (int kk = 0; kk < 64; kk += 32) {   // O += P @ V
      v16bf a  = frag_a(&Ps[wr * 16][kk], 64, lane);
      v16bf b0 = frag_b(&Vs[wc * 32][kk], 64, lane);
      v16bf b1 = frag_b(&Vs[wc * 32 + 16][kk], 64, lane);
      o0 = wmma_bf16(a, b0, o0);
      o1 = wmma_bf16(a, b1, o1);
    }
  }
  __syncthreads();
#pragma unroll
  for (int j = 0; j < 8; ++j) {
    float inv = 1.f / l_s[sm + j];
    O[base + (size_t)(q0 + sm + j) * DD + sn]      = o0[j] * inv;
    O[base + (size_t)(q0 + sm + j) * DD + sn + 16] = o1[j] * inv;
  }
}

// ---------------------------------------------------------------------------
// h = LayerNorm(x + o_proj)*g + b.  Writes fp32 h (gate/ref path) + bf16 shadow.
// ---------------------------------------------------------------------------
__global__ __launch_bounds__(256) void ln_kernel(
    const float* __restrict__ x, const float* __restrict__ o,
    const float* __restrict__ g, const float* __restrict__ beta,
    float* __restrict__ out, __bf16* __restrict__ out_bf) {
  __shared__ float red[256];
  __shared__ float s_mu, s_rstd;
  const int row = blockIdx.x;
  const int t = threadIdx.x;
  const float* xr = x + (size_t)row * DD;
  const float* orr = o + (size_t)row * DD;
  float v0 = xr[t] + orr[t];
  float v1 = xr[t + 256] + orr[t + 256];
  red[t] = v0 + v1;
  __syncthreads();
  for (int s = 128; s > 0; s >>= 1) {
    if (t < s) red[t] += red[t + s];
    __syncthreads();
  }
  if (t == 0) s_mu = red[0] * (1.f / 512.f);
  __syncthreads();
  float mu = s_mu;
  float d0 = v0 - mu, d1 = v1 - mu;
  red[t] = d0 * d0 + d1 * d1;
  __syncthreads();
  for (int s = 128; s > 0; s >>= 1) {
    if (t < s) red[t] += red[t + s];
    __syncthreads();
  }
  if (t == 0) s_rstd = rsqrtf(red[0] * (1.f / 512.f) + 1e-5f);
  __syncthreads();
  float rs = s_rstd;
  float h0 = d0 * rs * g[t] + beta[t];
  float h1 = d1 * rs * g[t + 256] + beta[t + 256];
  out[(size_t)row * DD + t]       = h0;
  out[(size_t)row * DD + t + 256] = h1;
  out_bf[(size_t)row * DD + t]       = (__bf16)h0;
  out_bf[(size_t)row * DD + t + 256] = (__bf16)h1;
}

// ---------------------------------------------------------------------------
// Top-1 gating: one wave per token (fp32 h). softmax max-prob = 1/sum(exp(l-lmax)).
// ---------------------------------------------------------------------------
__global__ __launch_bounds__(256) void gate_kernel(
    const float* __restrict__ h, const float* __restrict__ gW,
    const float* __restrict__ gb, int* __restrict__ idx,
    float* __restrict__ gwv, int* __restrict__ counts) {
  const int t = threadIdx.x, lane = t & 31, w = t >> 5;
  const int n = blockIdx.x * 8 + w;
  const float* hr = h + (size_t)n * DD;
  float acc[NE];
#pragma unroll
  for (int e = 0; e < NE; ++e) acc[e] = 0.f;
  for (int d = lane; d < DD; d += 32) {
    float hv = hr[d];
#pragma unroll
    for (int e = 0; e < NE; ++e) acc[e] += hv * gW[d * NE + e];
  }
#pragma unroll
  for (int e = 0; e < NE; ++e)
    for (int off = 16; off > 0; off >>= 1)
      acc[e] += __shfl_xor(acc[e], off);
  if (lane == 0) {
    float mx = -1e30f; int best = 0;
#pragma unroll
    for (int e = 0; e < NE; ++e) {
      acc[e] += gb[e];
      if (acc[e] > mx) { mx = acc[e]; best = e; }
    }
    float sum = 0.f;
#pragma unroll
    for (int e = 0; e < NE; ++e) sum += __expf(acc[e] - mx);
    idx[n] = best;
    gwv[n] = 1.f / sum;
    atomicAdd(&counts[best], 1);
  }
}

__global__ void zero_kernel(int* p, int n) {
  int i = threadIdx.x;
  if (i < n) p[i] = 0;
}

// Prefix scan + per-expert tile table (single thread; trivial work).
__global__ void scan_kernel(const int* __restrict__ counts, int* __restrict__ offsets,
                            int* __restrict__ tileE, int* __restrict__ tileStart,
                            int* __restrict__ tileRows, int* __restrict__ nTiles) {
  int off = 0, tid = 0;
  for (int e = 0; e < NE; ++e) {
    offsets[e] = off;
    int c = counts[e];
    for (int s = 0; s < c; s += 64) {
      tileE[tid] = e;
      tileStart[tid] = off + s;
      tileRows[tid] = min(64, c - s);
      ++tid;
    }
    off += c;
  }
  offsets[NE] = off;
  *nTiles = tid;
}

__global__ __launch_bounds__(256) void scatter_kernel(
    const int* __restrict__ idx, const int* __restrict__ offsets,
    int* __restrict__ cursor, int* __restrict__ perm) {
  int n = blockIdx.x * 256 + threadIdx.x;
  int e = idx[n];
  int slot = atomicAdd(&cursor[e], 1);
  perm[offsets[e] + slot] = n;
}

// ---------------------------------------------------------------------------
// FFN1: hid[p,f] = gelu(h_bf[perm[p]] @ W1[e] + b1[e]); hid stored bf16.
// A tiles async-gathered from bf16 h; B tiles packed-cvt from fp32 W1.
// ---------------------------------------------------------------------------
__global__ __launch_bounds__(256) void ffn1_kernel(
    const __bf16* __restrict__ hbf, const float* __restrict__ W1,
    const float* __restrict__ b1, const int* __restrict__ perm,
    const int* __restrict__ tileE, const int* __restrict__ tileStart,
    const int* __restrict__ tileRows, const int* __restrict__ nTiles,
    __bf16* __restrict__ hid) {
  const int tl = blockIdx.x;
  if (tl >= *nTiles) return;                 // uniform per block
  __shared__ __align__(16) __bf16 As[64][32];
  __shared__ __align__(16) __bf16 Bs[64][32];
  __shared__ int tok[64];
  const int e = tileE[tl], pstart = tileStart[tl], nrows = tileRows[tl];
  const int t = threadIdx.x, lane = t & 31, w = t >> 5;
  const int wr = w >> 1, wc = w & 1;
  if (t < 64) tok[t] = perm[pstart + min(t, nrows - 1)];
  const int f0 = blockIdx.y * 64;
  const float* W1e = W1 + (size_t)e * DD * FFD;
  const int arow = t >> 2, aseg = t & 3;     // A tile: 64 rows x 64B = 256 x 16B
  v8f c0 = {}; v8f c1 = {};
  for (int k0 = 0; k0 < DD; k0 += 32) {
    __syncthreads();
    async_ld_b128(&As[arow][aseg * 8],
                  hbf + (size_t)tok[arow] * DD + k0 + aseg * 8);
#pragma unroll
    for (int r = 0; r < 4; ++r) {            // W1 tile 32x64 -> Bs[n][k]
      int i = t + r * 256;
      int n = i & 63, kp = i >> 6;
      size_t gi = (size_t)(k0 + kp * 2) * FFD + f0 + n;
      ((uint32_t*)Bs)[n * 16 + kp] = pack_bf16(W1e[gi], W1e[gi + FFD]);
    }
    wait_async0();
    __syncthreads();
    v16bf a  = frag_a(&As[wr * 16][0], 32, lane);
    v16bf b0 = frag_b(&Bs[wc * 32][0], 32, lane);
    v16bf b1 = frag_b(&Bs[wc * 32 + 16][0], 32, lane);
    c0 = wmma_bf16(a, b0, c0);
    c1 = wmma_bf16(a, b1, c1);
  }
  const int nn = wc * 32 + (lane & 15);
  const int mm = wr * 16 + ((lane >> 4) << 3);
#pragma unroll
  for (int j = 0; j < 8; ++j) {
    int m = mm + j;
    if (m < nrows) {
      float v0 = c0[j] + b1[e * FFD + f0 + nn];
      float v1 = c1[j] + b1[e * FFD + f0 + nn + 16];
      hid[(size_t)(pstart + m) * FFD + f0 + nn]      = (__bf16)gelu_tanh(v0);
      hid[(size_t)(pstart + m) * FFD + f0 + nn + 16] = (__bf16)gelu_tanh(v1);
    }
  }
}

// ---------------------------------------------------------------------------
// FFN2: out[tok[p], d] = gw[tok[p]] * (hid[p] @ W2[e] + b2[e]).
// A tiles async-copied from bf16 hid.
// ---------------------------------------------------------------------------
__global__ __launch_bounds__(256) void ffn2_kernel(
    const __bf16* __restrict__ hid, const float* __restrict__ W2,
    const float* __restrict__ b2, const int* __restrict__ perm,
    const float* __restrict__ gwv,
    const int* __restrict__ tileE, const int* __restrict__ tileStart,
    const int* __restrict__ tileRows, const int* __restrict__ nTiles,
    float* __restrict__ out) {
  const int tl = blockIdx.x;
  if (tl >= *nTiles) return;
  __shared__ __align__(16) __bf16 As[64][32];
  __shared__ __align__(16) __bf16 Bs[64][32];
  __shared__ int tok[64];
  __shared__ float gws[64];
  const int e = tileE[tl], pstart = tileStart[tl], nrows = tileRows[tl];
  const int t = threadIdx.x, lane = t & 31, w = t >> 5;
  const int wr = w >> 1, wc = w & 1;
  if (t < 64) {
    int p = pstart + min(t, nrows - 1);
    int n = perm[p];
    tok[t] = n;
    gws[t] = gwv[n];
  }
  const int d0 = blockIdx.y * 64;
  const float* W2e = W2 + (size_t)e * FFD * DD;
  const int arow = t >> 2, aseg = t & 3;
  const size_t abase = (size_t)(pstart + min(arow, nrows - 1)) * FFD + aseg * 8;
  v8f c0 = {}; v8f c1 = {};
  for (int k0 = 0; k0 < FFD; k0 += 32) {
    __syncthreads();
    async_ld_b128(&As[arow][aseg * 8], hid + abase + k0);
#pragma unroll
    for (int r = 0; r < 4; ++r) {            // W2 tile 32x64 -> Bs[n][k]
      int i = t + r * 256;
      int n = i & 63, kp = i >> 6;
      size_t gi = (size_t)(k0 + kp * 2) * DD + d0 + n;
      ((uint32_t*)Bs)[n * 16 + kp] = pack_bf16(W2e[gi], W2e[gi + DD]);
    }
    wait_async0();
    __syncthreads();
    v16bf a  = frag_a(&As[wr * 16][0], 32, lane);
    v16bf b0 = frag_b(&Bs[wc * 32][0], 32, lane);
    v16bf b1 = frag_b(&Bs[wc * 32 + 16][0], 32, lane);
    c0 = wmma_bf16(a, b0, c0);
    c1 = wmma_bf16(a, b1, c1);
  }
  const int nn = wc * 32 + (lane & 15);
  const int mm = wr * 16 + ((lane >> 4) << 3);
#pragma unroll
  for (int j = 0; j < 8; ++j) {
    int m = mm + j;
    if (m < nrows) {
      float sc = gws[m];
      out[(size_t)tok[m] * DD + d0 + nn]      = (c0[j] + b2[e * DD + d0 + nn]) * sc;
      out[(size_t)tok[m] * DD + d0 + nn + 16] = (c1[j] + b2[e * DD + d0 + nn + 16]) * sc;
    }
  }
}

// ---------------------------------------------------------------------------
extern "C" void kernel_launch(void* const* d_in, const int* in_sizes, int n_in,
                              void* d_out, int out_size, void* d_ws, size_t ws_size,
                              hipStream_t stream) {
  (void)in_sizes; (void)n_in; (void)out_size; (void)ws_size;
  const float* x     = (const float*)d_in[0];
  const float* ln_g  = (const float*)d_in[1];
  const float* ln_b  = (const float*)d_in[2];
  const float* Wq    = (const float*)d_in[3];
  const float* bq    = (const float*)d_in[4];
  const float* Wk    = (const float*)d_in[5];
  const float* bk    = (const float*)d_in[6];
  const float* Wv    = (const float*)d_in[7];
  const float* bv    = (const float*)d_in[8];
  const float* Wo    = (const float*)d_in[9];
  const float* bo    = (const float*)d_in[10];
  const float* gW    = (const float*)d_in[11];
  const float* gb    = (const float*)d_in[12];
  const float* W1    = (const float*)d_in[13];
  const float* b1    = (const float*)d_in[14];
  const float* W2    = (const float*)d_in[15];
  const float* b2    = (const float*)d_in[16];
  float* out = (float*)d_out;

  // workspace layout (bf16 buffers first; every segment at least 4B-aligned)
  const size_t ND = (size_t)NB * DD;
  __bf16* Qb  = (__bf16*)d_ws;
  __bf16* Kb  = Qb + ND;
  __bf16* Vb  = Kb + ND;
  __bf16* Hbf = Vb + ND;
  __bf16* hid = Hbf + ND;               // NB*FFD bf16
  float*  AO  = (float*)(hid + (size_t)NB * FFD);
  float*  OP  = AO + ND;
  float*  Hb  = OP + ND;
  char* p = (char*)(Hb + ND);
  int*   idx     = (int*)p;          p += NB * sizeof(int);
  float* gwv     = (float*)p;        p += NB * sizeof(float);
  int*   perm    = (int*)p;          p += NB * sizeof(int);
  int*   counts  = (int*)p;          p += NE * sizeof(int);
  int*   cursor  = (int*)p;          p += NE * sizeof(int);
  int*   offsets = (int*)p;          p += (NE + 1) * sizeof(int);
  int*   tileE   = (int*)p;          p += MAXTILES * sizeof(int);
  int*   tileS   = (int*)p;          p += MAXTILES * sizeof(int);
  int*   tileR   = (int*)p;          p += MAXTILES * sizeof(int);
  int*   nTiles  = (int*)p;          p += sizeof(int);

  const dim3 gproj(NB / 64, DD / 64);   // 128 x 8

  // Q,K,V projections -> bf16
  gemm_kernel<__bf16><<<gproj, 256, 0, stream>>>(x, Wq, bq, Qb, NB, DD, DD);
  gemm_kernel<__bf16><<<gproj, 256, 0, stream>>>(x, Wk, bk, Kb, NB, DD, DD);
  gemm_kernel<__bf16><<<gproj, 256, 0, stream>>>(x, Wv, bv, Vb, NB, DD, DD);

  // flash attention (bf16 in, fp32 out)
  flash_kernel<<<dim3(SS / 64, 4 * NH), 256, 0, stream>>>(Qb, Kb, Vb, AO);

  // output projection + residual/LN (fp32 h + bf16 shadow)
  gemm_kernel<float><<<gproj, 256, 0, stream>>>(AO, Wo, bo, OP, NB, DD, DD);
  ln_kernel<<<NB, 256, 0, stream>>>(x, OP, ln_g, ln_b, Hb, Hbf);

  // gating + routing
  zero_kernel<<<1, 32, 0, stream>>>(counts, 2 * NE);   // counts+cursor adjacent
  gate_kernel<<<NB / 8, 256, 0, stream>>>(Hb, gW, gb, idx, gwv, counts);
  scan_kernel<<<1, 1, 0, stream>>>(counts, offsets, tileE, tileS, tileR, nTiles);
  scatter_kernel<<<NB / 256, 256, 0, stream>>>(idx, offsets, cursor, perm);

  // expert FFN (fixed worst-case tile grid; extra blocks exit via *nTiles)
  ffn1_kernel<<<dim3(NB / 64 + NE, FFD / 64), 256, 0, stream>>>(
      Hbf, W1, b1, perm, tileE, tileS, tileR, nTiles, hid);
  ffn2_kernel<<<dim3(NB / 64 + NE, DD / 64), 256, 0, stream>>>(
      hid, W2, b2, perm, gwv, tileE, tileS, tileR, nTiles, out);
}
